// NMN_1915555414394
// MI455X (gfx1250) — matrix operand order repaired and checked by reading, not compiled, not run on hardware
//
#include <hip/hip_runtime.h>
#include <math.h>

// ---- problem dims (match reference) ----
#define B_    128
#define C_    512
#define HW_   196      // 14*14
#define KF_   2
#define NROOT_ 64
#define A_    2000
#define AP_   2048     // A padded to /16 for WMMA tiles
#define T_    20
#define E_    300
#define EP_   320      // E padded to /32 for WMMA K
#define LH_   1024
#define Z4_   4096     // 4*LH
#define MH_   512

typedef unsigned short u16;
typedef __attribute__((ext_vector_type(16))) __bf16 v16bf;
typedef __attribute__((ext_vector_type(8)))  float  v8f;

union FragBF { uint4 u[2]; v16bf v; };

__device__ __forceinline__ u16 f2bf(float f) {
  unsigned int x = __float_as_uint(f);
  x += 0x7FFFu + ((x >> 16) & 1u);        // round-to-nearest-even
  return (u16)(x >> 16);
}

// ---------------------------------------------------------------------------
// Pack: src [Ksrc, Nsrc] f32 row-major  ->  dst [Npad, Kpad] bf16 row-major
// (transposed, zero-padded). Gives WMMA B fragments contiguous K runs.
// ---------------------------------------------------------------------------
__global__ void k_pack_T(const float* __restrict__ src, u16* __restrict__ dst,
                         int Ksrc, int Nsrc, int Kpad, int Npad) {
  int idx = blockIdx.x * blockDim.x + threadIdx.x;
  if (idx >= Kpad * Npad) return;
  int n = idx / Kpad, k = idx % Kpad;
  float v = (k < Ksrc && n < Nsrc) ? src[(size_t)k * Nsrc + n] : 0.f;
  dst[idx] = f2bf(v);
}

// Embedding gather + bf16 + pad: embX [T, B, EP] bf16
__global__ void k_pack_embx(const float* __restrict__ E_emb,
                            const int* __restrict__ question,
                            u16* __restrict__ embX) {
  int idx = blockIdx.x * blockDim.x + threadIdx.x;
  if (idx >= T_ * B_ * EP_) return;
  int k = idx % EP_;
  int rem = idx / EP_;
  int b = rem % B_, t = rem / B_;
  int q = question[b * T_ + t];
  float v = (k < E_) ? E_emb[(size_t)q * E_ + k] : 0.f;
  embX[idx] = f2bf(v);
}

__global__ void k_init_state(u16* __restrict__ h_bf, float* __restrict__ c) {
  int idx = blockIdx.x * blockDim.x + threadIdx.x;
  if (idx >= B_ * LH_) return;
  h_bf[idx] = 0; c[idx] = 0.f;
}

// ---------------------------------------------------------------------------
// bf16 WMMA GEMM, 2x2 register-blocked: each wave computes a 32x32 D tile.
// D[M,N] = A1[M,K1]*B1[N,K1]^T + A2[M,K2]*B2[N,K2]^T
// A row-major [M,K], B stored transposed [N,K] (bf16, K%32==0, M,N%32==0).
// Fragment layouts per CDNA5 ISA 7.12.2:
//   A: lane(m=l&15, g=l>>4) holds K runs [g*8, g*8+8) and [16+g*8, ...)
//   B: lane(n=l&15, g=l>>4) holds K runs [g*8, g*8+8) and [16+g*8, ...), same as A
//   C/D: VGPR r -> (m = r + 8*g, n = l&15)
// 8 b128 loads feed 4 WMMAs per K-tile (2 loads/WMMA).
// ---------------------------------------------------------------------------
__device__ __forceinline__ void wmma_seg(const u16* __restrict__ A, const u16* __restrict__ B,
                                         int K, size_t ra0, size_t ra1,
                                         size_t rb0, size_t rb1, int g8, int g16,
                                         v8f& c00, v8f& c01, v8f& c10, v8f& c11) {
  const u16* a0 = A + ra0 * K;
  const u16* a1 = A + ra1 * K;
  const u16* b0 = B + rb0 * K;
  const u16* b1 = B + rb1 * K;
  for (int k0 = 0; k0 < K; k0 += 32) {
    FragBF fa0, fa1, fb0, fb1;
    fa0.u[0] = *(const uint4*)(a0 + k0 + g8);
    fa0.u[1] = *(const uint4*)(a0 + k0 + 16 + g8);
    fa1.u[0] = *(const uint4*)(a1 + k0 + g8);
    fa1.u[1] = *(const uint4*)(a1 + k0 + 16 + g8);
    fb0.u[0] = *(const uint4*)(b0 + k0 + g8);
    fb0.u[1] = *(const uint4*)(b0 + k0 + 16 + g8);
    fb1.u[0] = *(const uint4*)(b1 + k0 + g8);
    fb1.u[1] = *(const uint4*)(b1 + k0 + 16 + g8);
    c00 = __builtin_amdgcn_wmma_f32_16x16x32_bf16(false, fa0.v, false, fb0.v, (short)0, c00, false, false);
    c01 = __builtin_amdgcn_wmma_f32_16x16x32_bf16(false, fa0.v, false, fb1.v, (short)0, c01, false, false);
    c10 = __builtin_amdgcn_wmma_f32_16x16x32_bf16(false, fa1.v, false, fb0.v, (short)0, c10, false, false);
    c11 = __builtin_amdgcn_wmma_f32_16x16x32_bf16(false, fa1.v, false, fb1.v, (short)0, c11, false, false);
  }
}

__global__ void k_gemm_bf16_wmma(const u16* __restrict__ A1, const u16* __restrict__ B1, int K1,
                                 const u16* __restrict__ A2, const u16* __restrict__ B2, int K2,
                                 float* __restrict__ D, int ldd, int tilesN /* = N/32 */) {
  int wave = blockIdx.x * (blockDim.x >> 5) + (threadIdx.x >> 5);
  int lane = threadIdx.x & 31;
  int tm = wave / tilesN, tn = wave % tilesN;
  int l15 = lane & 15;
  int g   = lane >> 4;
  int g8  = g << 3, g16 = g << 4;
  size_t ra0 = (size_t)(tm * 32 + l15), ra1 = ra0 + 16;
  size_t rb0 = (size_t)(tn * 32 + l15), rb1 = rb0 + 16;
  v8f c00 = {0.f,0.f,0.f,0.f,0.f,0.f,0.f,0.f};
  v8f c01 = c00, c10 = c00, c11 = c00;
  wmma_seg(A1, B1, K1, ra0, ra1, rb0, rb1, g8, g16, c00, c01, c10, c11);
  if (K2 > 0)
    wmma_seg(A2, B2, K2, ra0, ra1, rb0, rb1, g8, g16, c00, c01, c10, c11);
  int col0 = tn * 32 + l15, col1 = col0 + 16;
  int row0 = tm * 32 + g * 8, row1 = row0 + 16;
#pragma unroll
  for (int r = 0; r < 8; ++r) {
    D[(size_t)(row0 + r) * ldd + col0] = c00[r];
    D[(size_t)(row0 + r) * ldd + col1] = c01[r];
    D[(size_t)(row1 + r) * ldd + col0] = c10[r];
    D[(size_t)(row1 + r) * ldd + col1] = c11[r];
  }
}

// ---------------------------------------------------------------------------
// LSTM pointwise gates: z -> (c,h); h stored bf16 for next step's GEMM
// ---------------------------------------------------------------------------
__global__ void k_lstm_point(const float* __restrict__ z, const float* __restrict__ b_lstm,
                             float* __restrict__ c, u16* __restrict__ h_bf,
                             const int* __restrict__ length, u16* __restrict__ hlast_bf,
                             int t) {
  int idx = blockIdx.x * blockDim.x + threadIdx.x;
  if (idx >= B_ * LH_) return;
  int b = idx / LH_, j = idx % LH_;
  const float* zb = z + (size_t)b * Z4_;
  float zi = zb[j]           + b_lstm[j];
  float zf = zb[LH_ + j]     + b_lstm[LH_ + j];
  float zg = zb[2 * LH_ + j] + b_lstm[2 * LH_ + j];
  float zo = zb[3 * LH_ + j] + b_lstm[3 * LH_ + j];
  float si = 1.f / (1.f + __expf(-zi));
  float sf = 1.f / (1.f + __expf(-zf));
  float so = 1.f / (1.f + __expf(-zo));
  float cn = sf * c[idx] + si * tanhf(zg);
  float hn = so * tanhf(cn);
  c[idx] = cn;
  h_bf[idx] = f2bf(hn);
  if (t == length[b] - 1) hlast_bf[idx] = f2bf(hn);
}

// ---------------------------------------------------------------------------
// Find module: maps = relu(w0.F)*relu(w1.F); attended = (maps.F)/sum(maps)
// One block per sample; wave32 shuffle reductions.
// ---------------------------------------------------------------------------
__global__ void k_find(const float* __restrict__ features, const float* __restrict__ W_find,
                       const int* __restrict__ find_inst,
                       float* __restrict__ maps, float* __restrict__ attended) {
  int b = blockIdx.x, tid = threadIdx.x;
  __shared__ float w0[C_], w1[C_];
  __shared__ float smap[HW_];
  __shared__ float red[256];
  int r0 = find_inst[b * KF_ + 0], r1 = find_inst[b * KF_ + 1];
  for (int i = tid; i < C_; i += 256) {
    w0[i] = W_find[(size_t)r0 * C_ + i];
    w1[i] = W_find[(size_t)r1 * C_ + i];
  }
  __syncthreads();
  const float* fb = features + (size_t)b * C_ * HW_;
  if (tid < HW_) {
    float a0 = 0.f, a1 = 0.f;
    for (int cc = 0; cc < C_; ++cc) {
      float f = fb[cc * HW_ + tid];           // coalesced over hw
      a0 += w0[cc] * f; a1 += w1[cc] * f;
    }
    float m = fmaxf(a0, 0.f) * fmaxf(a1, 0.f);
    smap[tid] = m;
    maps[(size_t)b * HW_ + tid] = m;
  }
  __syncthreads();
  red[tid] = (tid < HW_) ? smap[tid] : 0.f;
  __syncthreads();
  for (int off = 128; off > 0; off >>= 1) {
    if (tid < off) red[tid] += red[tid + off];
    __syncthreads();
  }
  float denom = red[0] + 1e-6f;
  int wv = tid >> 5, lane = tid & 31;
  for (int cc = wv; cc < C_; cc += 8) {        // 8 waves, 64 channels each
    float acc = 0.f;
    for (int hw = lane; hw < HW_; hw += 32) acc += smap[hw] * fb[cc * HW_ + hw];
    for (int off = 16; off > 0; off >>= 1) acc += __shfl_down(acc, off, 32);
    if (lane == 0) attended[(size_t)b * C_ + cc] = acc / denom;
  }
}

// Measure MLP layer 1 (only for yesno samples)
__global__ void k_measure(const float* __restrict__ maps, const float* __restrict__ W_meas1,
                          const float* __restrict__ b_meas1,
                          const unsigned char* __restrict__ yesno,
                          const int* __restrict__ root_inst, float* __restrict__ h_meas) {
  int b = blockIdx.x, tid = threadIdx.x;
  if (!yesno[b]) return;                       // uniform per block
  int r = root_inst[b];
  __shared__ float sm[HW_];
  if (tid < HW_) sm[tid] = maps[(size_t)b * HW_ + tid];
  __syncthreads();
  const float* W = W_meas1 + (size_t)r * HW_ * MH_;
  for (int m = tid; m < MH_; m += 256) {
    float acc = b_meas1[(size_t)r * MH_ + m];
    for (int i = 0; i < HW_; ++i) acc += sm[i] * W[(size_t)i * MH_ + m];
    h_meas[(size_t)b * MH_ + m] = fmaxf(acc, 0.f);
  }
}

// Selected-expert logits + softmax -> root_pred. Only one branch per sample.
__global__ void k_root_pred(const float* __restrict__ h_meas, const float* __restrict__ attended,
                            const float* __restrict__ W_meas2, const float* __restrict__ b_meas2,
                            const float* __restrict__ W_desc, const float* __restrict__ b_desc,
                            const unsigned char* __restrict__ yesno,
                            const int* __restrict__ root_inst,
                            float* __restrict__ root_pred) {
  int b = blockIdx.x, tid = threadIdx.x;
  int yn = yesno[b] ? 1 : 0;
  int r = root_inst[b];
  __shared__ float vec[MH_];
  __shared__ float lg[A_];
  __shared__ float red[256];
  const float* src = yn ? (h_meas + (size_t)b * MH_) : (attended + (size_t)b * C_);
  for (int i = tid; i < MH_; i += 256) vec[i] = src[i];
  __syncthreads();
  if (yn) {
    for (int a = tid; a < A_; a += 256) {
      float acc = b_meas2[a];
      for (int m = 0; m < MH_; ++m) acc += vec[m] * W_meas2[(size_t)m * A_ + a];
      lg[a] = acc;
    }
  } else {
    const float* W = W_desc + (size_t)r * C_ * A_;   // stream: contiguous over a
    for (int a = tid; a < A_; a += 256) {
      float acc = b_desc[(size_t)r * A_ + a];
      for (int cb = 0; cb < C_; cb += 64) {
        // keep the 23.3 TB/s stream ahead of the FMAs (global_prefetch path)
        if (cb + 64 < C_) __builtin_prefetch(W + (size_t)(cb + 64) * A_ + a, 0, 0);
        for (int cc = cb; cc < cb + 64; ++cc) acc += vec[cc] * W[(size_t)cc * A_ + a];
      }
      lg[a] = acc;
    }
  }
  __syncthreads();
  float mx = -INFINITY;
  for (int a = tid; a < A_; a += 256) mx = fmaxf(mx, lg[a]);
  red[tid] = mx; __syncthreads();
  for (int off = 128; off > 0; off >>= 1) {
    if (tid < off) red[tid] = fmaxf(red[tid], red[tid + off]);
    __syncthreads();
  }
  mx = red[0]; __syncthreads();
  float sum = 0.f;
  for (int a = tid; a < A_; a += 256) { float e = __expf(lg[a] - mx); lg[a] = e; sum += e; }
  red[tid] = sum; __syncthreads();
  for (int off = 128; off > 0; off >>= 1) {
    if (tid < off) red[tid] += red[tid + off];
    __syncthreads();
  }
  float inv = 1.f / red[0];
  for (int a = tid; a < A_; a += 256) root_pred[(size_t)b * A_ + a] = lg[a] * inv;
}

// Encoder softmax + geometric blend -> output
__global__ void k_final(const float* __restrict__ z_enc, const float* __restrict__ b_enc,
                        const float* __restrict__ root_pred, float* __restrict__ out) {
  int b = blockIdx.x, tid = threadIdx.x;
  __shared__ float lg[A_];
  __shared__ float red[256];
  for (int a = tid; a < A_; a += 256) lg[a] = z_enc[(size_t)b * AP_ + a] + b_enc[a];
  __syncthreads();
  float mx = -INFINITY;
  for (int a = tid; a < A_; a += 256) mx = fmaxf(mx, lg[a]);
  red[tid] = mx; __syncthreads();
  for (int off = 128; off > 0; off >>= 1) {
    if (tid < off) red[tid] = fmaxf(red[tid], red[tid + off]);
    __syncthreads();
  }
  mx = red[0]; __syncthreads();
  float sum = 0.f;
  for (int a = tid; a < A_; a += 256) { float e = __expf(lg[a] - mx); lg[a] = e; sum += e; }
  red[tid] = sum; __syncthreads();
  for (int off = 128; off > 0; off >>= 1) {
    if (tid < off) red[tid] += red[tid + off];
    __syncthreads();
  }
  float inv = 1.f / red[0];
  for (int a = tid; a < A_; a += 256)
    out[(size_t)b * A_ + a] = sqrtf(root_pred[(size_t)b * A_ + a] * lg[a] * inv);
}

// ---------------------------------------------------------------------------
extern "C" void kernel_launch(void* const* d_in, const int* in_sizes, int n_in,
                              void* d_out, int out_size, void* d_ws, size_t ws_size,
                              hipStream_t stream) {
  (void)in_sizes; (void)n_in; (void)out_size; (void)ws_size;
  const float* features  = (const float*)d_in[0];
  const int*   question  = (const int*)d_in[1];
  const int*   length    = (const int*)d_in[2];
  const unsigned char* yesno = (const unsigned char*)d_in[3];   // jax bool = 1 byte
  const int*   root_inst = (const int*)d_in[4];
  const int*   find_inst = (const int*)d_in[5];
  const float* W_find  = (const float*)d_in[6];
  const float* W_meas1 = (const float*)d_in[7];
  const float* b_meas1 = (const float*)d_in[8];
  const float* W_meas2 = (const float*)d_in[9];
  const float* b_meas2 = (const float*)d_in[10];
  const float* W_desc  = (const float*)d_in[11];
  const float* b_desc  = (const float*)d_in[12];
  const float* E_emb   = (const float*)d_in[13];
  const float* Wx      = (const float*)d_in[14];
  const float* Wh      = (const float*)d_in[15];
  const float* b_lstm  = (const float*)d_in[16];
  const float* W_enc   = (const float*)d_in[17];
  const float* b_enc   = (const float*)d_in[18];
  float* out = (float*)d_out;

  char* ws = (char*)d_ws;
  size_t off = 0;
  auto alloc = [&](size_t bytes) -> void* {
    void* p = (void*)(ws + off);
    off += (bytes + 255) & ~(size_t)255;
    return p;
  };
  float* maps_ws     = (float*)alloc((size_t)B_ * HW_ * 4);
  float* attended_ws = (float*)alloc((size_t)B_ * C_ * 4);
  float* hmeas_ws    = (float*)alloc((size_t)B_ * MH_ * 4);
  float* rootpred_ws = (float*)alloc((size_t)B_ * A_ * 4);
  u16*   embX        = (u16*)  alloc((size_t)T_ * B_ * EP_ * 2);
  u16*   WxT         = (u16*)  alloc((size_t)Z4_ * EP_ * 2);
  u16*   WhT         = (u16*)  alloc((size_t)Z4_ * LH_ * 2);
  u16*   WencT       = (u16*)  alloc((size_t)AP_ * LH_ * 2);
  u16*   h_bf        = (u16*)  alloc((size_t)B_ * LH_ * 2);
  u16*   hlast_bf    = (u16*)  alloc((size_t)B_ * LH_ * 2);
  float* c_ws        = (float*)alloc((size_t)B_ * LH_ * 4);
  float* z_ws        = (float*)alloc((size_t)B_ * Z4_ * 4);
  float* zenc_ws     = (float*)alloc((size_t)B_ * AP_ * 4);

  // --- pack weights/embeddings to bf16, transposed [N,K] ---
  k_pack_T<<<(LH_ * Z4_ + 255) / 256, 256, 0, stream>>>(Wh, WhT, LH_, Z4_, LH_, Z4_);
  k_pack_T<<<(EP_ * Z4_ + 255) / 256, 256, 0, stream>>>(Wx, WxT, E_, Z4_, EP_, Z4_);
  k_pack_T<<<(LH_ * AP_ + 255) / 256, 256, 0, stream>>>(W_enc, WencT, LH_, A_, LH_, AP_);
  k_pack_embx<<<(T_ * B_ * EP_ + 255) / 256, 256, 0, stream>>>(E_emb, question, embX);
  k_init_state<<<(B_ * LH_ + 255) / 256, 256, 0, stream>>>(h_bf, c_ws);

  // --- module network (memory-bound, selected expert only) ---
  k_find<<<B_, 256, 0, stream>>>(features, W_find, find_inst, maps_ws, attended_ws);
  k_measure<<<B_, 256, 0, stream>>>(maps_ws, W_meas1, b_meas1, yesno, root_inst, hmeas_ws);
  k_root_pred<<<B_, 256, 0, stream>>>(hmeas_ws, attended_ws, W_meas2, b_meas2,
                                      W_desc, b_desc, yesno, root_inst, rootpred_ws);

  // --- LSTM: 20 sequential (WMMA GEMM + gates) steps; x-proj folded as K-ext ---
  const int tilesN_z = Z4_ / 32;                 // 128 N-tiles of 32
  const int waves_z  = (B_ / 32) * tilesN_z;     // 512 waves -> 128 blocks of 4 waves
  for (int t = 0; t < T_; ++t) {
    k_gemm_bf16_wmma<<<waves_z / 4, 128, 0, stream>>>(
        h_bf, WhT, LH_, embX + (size_t)t * B_ * EP_, WxT, EP_, z_ws, Z4_, tilesN_z);
    k_lstm_point<<<(B_ * LH_ + 255) / 256, 256, 0, stream>>>(
        z_ws, b_lstm, c_ws, h_bf, length, hlast_bf, t);
  }

  // --- encoder head GEMM (WMMA) + softmax + blend ---
  const int tilesN_e = AP_ / 32;                 // 64
  const int waves_e  = (B_ / 32) * tilesN_e;     // 256 waves -> 64 blocks
  k_gemm_bf16_wmma<<<waves_e / 4, 128, 0, stream>>>(
      hlast_bf, WencT, LH_, (const u16*)nullptr, (const u16*)nullptr, 0,
      zenc_ws, AP_, tilesN_e);
  k_final<<<B_, 256, 0, stream>>>(zenc_ws, b_enc, rootpred_ws, out);
}
